// IGAP_56530359549977
// MI455X (gfx1250) — matrix operand order repaired and checked by reading, not compiled, not run on hardware
//
#include <hip/hip_runtime.h>
#include <hip/hip_bf16.h>
#include <stdint.h>

#define NN 100000
#define NE 1600000
#define IN_CH 256
#define PROJ 128
#define HID 128
#define OUT_CH 32

typedef __attribute__((ext_vector_type(16))) __bf16 v16bf;
typedef __attribute__((ext_vector_type(8)))  float  v8f;

union FragBF { v16bf v; uint4 q[2]; };

__device__ __forceinline__ unsigned short f2bf(float f) {
  unsigned int u = __float_as_uint(f);
  unsigned int r = 0x7FFFu + ((u >> 16) & 1u);   // round-to-nearest-even
  return (unsigned short)((u + r) >> 16);
}

// ---------------- elementwise / graph kernels ----------------

__global__ void cast_bf16_k(const float* __restrict__ in, unsigned short* __restrict__ out, int n4) {
  int i = blockIdx.x * blockDim.x + threadIdx.x;
  if (i >= n4) return;
  float4 v = *(const float4*)(in + (size_t)i * 4);
  size_t b = (size_t)i * 4;
  out[b + 0] = f2bf(v.x); out[b + 1] = f2bf(v.y);
  out[b + 2] = f2bf(v.z); out[b + 3] = f2bf(v.w);
}

// Pack row-major f32 weights [K,N] into WMMA B-fragment order (bf16).
// dword index: frag*256 + lane*8 + v ; frag = kt*NT + j
// lane group g = lane>>4 covers K 8g..8g+7 and 16+8g..16+8g+7 ; dword v holds K = kt*32 + 8g + 2v + (v>=4 ? 8 : 0) (lo), +1 (hi)
__global__ void pack_w_k(const float* __restrict__ W, unsigned int* __restrict__ out, int K, int N) {
  int idx = blockIdx.x * blockDim.x + threadIdx.x;
  int NT = N >> 4;
  int total = (K >> 5) * NT * 256;
  if (idx >= total) return;
  int v = idx & 7;
  int lane = (idx >> 3) & 31;
  int f = idx >> 8;
  int j = f % NT;
  int kt = f / NT;
  int g = lane >> 4;
  int n = j * 16 + (lane & 15);
  int k = kt * 32 + 8 * g + 2 * v + (v >= 4 ? 8 : 0);
  unsigned int lo = f2bf(W[(size_t)k * N + n]);
  unsigned int hi = f2bf(W[(size_t)(k + 1) * N + n]);
  out[idx] = lo | (hi << 16);
}

__global__ void deg_init_k(float* __restrict__ deg, int n) {
  int i = blockIdx.x * blockDim.x + threadIdx.x;
  if (i < n) deg[i] = 1.0f;                      // self-loop weight
}

__global__ void deg_acc_k(const int* __restrict__ dst, const float* __restrict__ ew,
                          float* __restrict__ deg, int E) {
  int e = blockIdx.x * blockDim.x + threadIdx.x;
  if (e < E) atomicAdd(&deg[dst[e]], ew[e]);
}

__global__ void dinv_k(const float* __restrict__ deg, float* __restrict__ dinv, int n) {
  int i = blockIdx.x * blockDim.x + threadIdx.x;
  if (i < n) dinv[i] = rsqrtf(deg[i]);
}

__global__ void norm_k(const int* __restrict__ src, const int* __restrict__ dst,
                       const float* __restrict__ ew, const float* __restrict__ dinv,
                       float* __restrict__ norm, int E) {
  int e = blockIdx.x * blockDim.x + threadIdx.x;
  if (e < E) norm[e] = dinv[src[e]] * ew[e] * dinv[dst[e]];
}

// agg[n][c] = dinv[n]^2 * hw[n][c]   (self-loop message; also initializes agg)
__global__ void self_init_k(const float* __restrict__ hw, const float* __restrict__ dinv,
                            float* __restrict__ agg, int n) {
  int idx = blockIdx.x * blockDim.x + threadIdx.x;   // n*32 threads, float4 each
  int node = idx >> 5;
  if (node >= n) return;
  int c = (idx & 31) << 2;
  float d = dinv[node];
  float dd = d * d;
  float4 v = *(const float4*)(hw + (size_t)node * HID + c);
  float4 o = make_float4(dd * v.x, dd * v.y, dd * v.z, dd * v.w);
  *(float4*)(agg + (size_t)node * HID + c) = o;
}

// agg[dst] += norm[e] * hw[src]  (float atomics; gathers hit 192MB L2)
__global__ void scatter_k(const int* __restrict__ src, const int* __restrict__ dst,
                          const float* __restrict__ norm, const float* __restrict__ hw,
                          float* __restrict__ agg, int E) {
  int idx = blockIdx.x * blockDim.x + threadIdx.x;   // E*32 threads, float4 each
  int e = idx >> 5;
  if (e >= E) return;
  int c = (idx & 31) << 2;
  int s = src[e], d = dst[e];
  float w = norm[e];
  float4 v = *(const float4*)(hw + (size_t)s * HID + c);
  float* o = agg + (size_t)d * HID + c;
  atomicAdd(o + 0, w * v.x);
  atomicAdd(o + 1, w * v.y);
  atomicAdd(o + 2, w * v.z);
  atomicAdd(o + 3, w * v.w);
}

// out_bf16[n][c] = bf16(relu(agg[n][c] + bias[c]))
__global__ void finalize_k(const float* __restrict__ agg, const float* __restrict__ bias,
                           unsigned short* __restrict__ out, int total) {
  int i = blockIdx.x * blockDim.x + threadIdx.x;
  if (i >= total) return;
  float v = agg[i] + bias[i & (HID - 1)];
  out[i] = f2bf(v > 0.0f ? v : 0.0f);
}

__global__ void softmax_k(const float* __restrict__ logits, float* __restrict__ out, int n) {
  int i = blockIdx.x * blockDim.x + threadIdx.x;
  if (i >= n) return;
  const float* r = logits + (size_t)i * OUT_CH;
  float m = r[0];
#pragma unroll
  for (int c = 1; c < OUT_CH; ++c) m = fmaxf(m, r[c]);
  float e[OUT_CH];
  float s = 0.0f;
#pragma unroll
  for (int c = 0; c < OUT_CH; ++c) { e[c] = __expf(r[c] - m); s += e[c]; }
  float inv = 1.0f / s;
  float* o = out + (size_t)i * OUT_CH;
#pragma unroll
  for (int c = 0; c < OUT_CH; ++c) o[c] = e[c] * inv;
}

// ---------------- WMMA GEMM ----------------
// C[M, NT*16] = A[M,K](bf16 row-major) x Bpacked(bf16 fragments) (+bias)(+relu)
// One wave per 16-row M tile, all NT column tiles accumulated in registers.
template<int NT, bool RELU, bool OUT_BF16>
__global__ __launch_bounds__(256)
void gemm_wmma_k(const unsigned short* __restrict__ A,
                 const unsigned int* __restrict__ Bp,
                 const float* __restrict__ bias,
                 void* __restrict__ outp,
                 int M, int K) {
  const int lane = threadIdx.x & 31;
  const int warp = threadIdx.x >> 5;
  const int mtile = blockIdx.x * 8 + warp;
  if (mtile * 16 >= M) return;                 // whole-wave exit; EXEC stays all-1s

  const int g  = lane >> 4;
  const int ln = lane & 15;
  const int row = mtile * 16 + ln;             // A row this lane feeds
  const int kb  = g * 8;

  const v8f zero = {0.f, 0.f, 0.f, 0.f, 0.f, 0.f, 0.f, 0.f};
  v8f acc[NT];
#pragma unroll
  for (int j = 0; j < NT; ++j) acc[j] = zero;

  const unsigned short* arow = A + (size_t)row * K;
  const uint4* bq = (const uint4*)Bp;

  for (int k0 = 0; k0 < K; k0 += 32) {
    FragBF a;
    a.q[0] = *(const uint4*)(arow + k0 + kb);        // K = k0+8g .. +7
    a.q[1] = *(const uint4*)(arow + k0 + kb + 16);   // K = k0+8g+16 .. +23
    const int kt = k0 >> 5;
#pragma unroll
    for (int j = 0; j < NT; ++j) {
      FragBF b;
      const uint4* p = bq + ((size_t)(kt * NT + j) * 32 + lane) * 2;
      b.q[0] = p[0];
      b.q[1] = p[1];
      acc[j] = __builtin_amdgcn_wmma_f32_16x16x32_bf16(
          false, a.v, false, b.v, (short)0, acc[j], false, false);
    }
  }

  const int Ntot = NT * 16;
#pragma unroll
  for (int j = 0; j < NT; ++j) {
    float bv = bias ? bias[j * 16 + ln] : 0.0f;
#pragma unroll
    for (int r = 0; r < 8; ++r) {
      float v = acc[j][r] + bv;
      if (RELU) v = v > 0.0f ? v : 0.0f;
      int m = mtile * 16 + r + 8 * g;          // C layout: M = r + 8g, N = ln
      int n = j * 16 + ln;
      if (OUT_BF16) ((unsigned short*)outp)[(size_t)m * Ntot + n] = f2bf(v);
      else          ((float*)outp)[(size_t)m * Ntot + n] = v;
    }
  }
}

// ---------------- launch ----------------

extern "C" void kernel_launch(void* const* d_in, const int* in_sizes, int n_in,
                              void* d_out, int out_size, void* d_ws, size_t ws_size,
                              hipStream_t stream) {
  const float* x   = (const float*)d_in[0];
  const int*   ei  = (const int*)  d_in[1];
  const float* ew  = (const float*)d_in[2];
  const float* Wp  = (const float*)d_in[3];
  const float* bp  = (const float*)d_in[4];
  const float* W1  = (const float*)d_in[5];
  const float* b1  = (const float*)d_in[6];
  const float* W2  = (const float*)d_in[7];
  const float* b2  = (const float*)d_in[8];
  const float* Wl1 = (const float*)d_in[9];
  const float* bl1 = (const float*)d_in[10];
  const float* Wl2 = (const float*)d_in[11];
  const float* bl2 = (const float*)d_in[12];
  const int* src = ei;
  const int* dst = ei + NE;

  char* ws = (char*)d_ws;
  size_t off = 0;
  auto alloc = [&](size_t bytes) -> void* {
    off = (off + 255) & ~(size_t)255;
    void* p = ws + off;
    off += bytes;
    return p;
  };

  unsigned short* xb   = (unsigned short*)alloc((size_t)NN * IN_CH * 2);
  unsigned short* hb   = (unsigned short*)alloc((size_t)NN * HID * 2);
  unsigned short* hb2  = (unsigned short*)alloc((size_t)NN * HID * 2);
  float*          hw   = (float*)alloc((size_t)NN * HID * 4);
  float*          agg  = (float*)alloc((size_t)NN * HID * 4);
  float*          deg  = (float*)alloc((size_t)NN * 4);
  float*          dinv = (float*)alloc((size_t)NN * 4);
  float*          nrm  = (float*)alloc((size_t)NE * 4);
  unsigned int*   wpP  = (unsigned int*)alloc((size_t)(IN_CH/32)*(PROJ/16)*256*4);
  unsigned int*   wp1  = (unsigned int*)alloc((size_t)(PROJ/32)*(HID/16)*256*4);
  unsigned int*   wp2  = (unsigned int*)alloc((size_t)(HID/32)*(HID/16)*256*4);
  unsigned int*   wpl1 = (unsigned int*)alloc((size_t)(HID/32)*(HID/16)*256*4);
  unsigned int*   wpl2 = (unsigned int*)alloc((size_t)(HID/32)*(OUT_CH/16)*256*4);

  const int TB = 256;
  auto blks = [](long long n, int tb) { return (int)((n + tb - 1) / tb); };

  // casts + weight packing (cheap, redone each call for determinism)
  cast_bf16_k<<<blks((long long)NN * IN_CH / 4, TB), TB, 0, stream>>>(x, xb, NN * IN_CH / 4);
  pack_w_k<<<blks((IN_CH/32)*(PROJ/16)*256, TB), TB, 0, stream>>>(Wp,  wpP,  IN_CH, PROJ);
  pack_w_k<<<blks((PROJ/32)*(HID/16)*256,  TB), TB, 0, stream>>>(W1,  wp1,  PROJ,  HID);
  pack_w_k<<<blks((HID/32)*(HID/16)*256,   TB), TB, 0, stream>>>(W2,  wp2,  HID,   HID);
  pack_w_k<<<blks((HID/32)*(HID/16)*256,   TB), TB, 0, stream>>>(Wl1, wpl1, HID,   HID);
  pack_w_k<<<blks((HID/32)*(OUT_CH/16)*256,TB), TB, 0, stream>>>(Wl2, wpl2, HID,   OUT_CH);

  // graph normalization (shared by both convs)
  deg_init_k<<<blks(NN, TB), TB, 0, stream>>>(deg, NN);
  deg_acc_k <<<blks(NE, TB), TB, 0, stream>>>(dst, ew, deg, NE);
  dinv_k    <<<blks(NN, TB), TB, 0, stream>>>(deg, dinv, NN);
  norm_k    <<<blks(NE, TB), TB, 0, stream>>>(src, dst, ew, dinv, nrm, NE);

  const int GEMM_BLKS = (NN / 16 + 7) / 8;   // 6250 M tiles, 8 waves per block

  // h0 = relu(x @ Wp + bp)  -> bf16
  gemm_wmma_k<PROJ/16, true, true><<<GEMM_BLKS, TB, 0, stream>>>(xb, wpP, bp, hb, NN, IN_CH);

  // conv1: hw = h0 @ W1 ; agg = dinv^2*hw + scatter(norm*hw[src]) ; h1 = relu(agg+b1)
  gemm_wmma_k<HID/16, false, false><<<GEMM_BLKS, TB, 0, stream>>>(hb, wp1, nullptr, hw, NN, PROJ);
  self_init_k<<<blks((long long)NN * 32, TB), TB, 0, stream>>>(hw, dinv, agg, NN);
  scatter_k  <<<blks((long long)NE * 32, TB), TB, 0, stream>>>(src, dst, nrm, hw, agg, NE);
  finalize_k <<<blks((long long)NN * HID, TB), TB, 0, stream>>>(agg, b1, hb, NN * HID);

  // conv2
  gemm_wmma_k<HID/16, false, false><<<GEMM_BLKS, TB, 0, stream>>>(hb, wp2, nullptr, hw, NN, HID);
  self_init_k<<<blks((long long)NN * 32, TB), TB, 0, stream>>>(hw, dinv, agg, NN);
  scatter_k  <<<blks((long long)NE * 32, TB), TB, 0, stream>>>(src, dst, nrm, hw, agg, NE);
  finalize_k <<<blks((long long)NN * HID, TB), TB, 0, stream>>>(agg, b2, hb, NN * HID);

  // h3 = relu(h2 @ Wl1 + bl1) -> bf16
  gemm_wmma_k<HID/16, true, true><<<GEMM_BLKS, TB, 0, stream>>>(hb, wpl1, bl1, hb2, NN, HID);

  // logits = h3 @ Wl2 + bl2 -> f32 (reuse hw)
  gemm_wmma_k<OUT_CH/16, false, false><<<GEMM_BLKS, TB, 0, stream>>>(hb2, wpl2, bl2, hw, NN, HID);

  // softmax -> d_out
  softmax_k<<<blks(NN, TB), TB, 0, stream>>>(hw, (float*)d_out, NN);
}
